// AttentionBlock_8452495638791
// MI455X (gfx1250) — matrix operand order repaired and checked by reading, not compiled, not run on hardware
//
#include <hip/hip_runtime.h>
#include <hip/hip_bf16.h>

// Problem constants (match reference)
#define BATCH 16
#define CCH   512
#define NSP   1024
#define NGRP  8
#define CPG   (CCH / NGRP)   // 64
#define O3C   (3 * CCH)      // 1536

typedef __attribute__((ext_vector_type(16))) __bf16 v16bf;
typedef __attribute__((ext_vector_type(8)))  __bf16 v8bf;
typedef __attribute__((ext_vector_type(8)))  float  v8f;
typedef unsigned short ushort_t;

__device__ __forceinline__ ushort_t f32_to_bf16(float f) {
  unsigned u = __float_as_uint(f);
  u += 0x7FFFu + ((u >> 16) & 1u);   // round to nearest even
  return (ushort_t)(u >> 16);
}

// ---- WMMA fragment loaders (per CDNA5 ISA 7.12.2 layouts, wave32) ----

// A-matrix 16x32 bf16, row-major source with leading dim lda (elements).
// lane<16: rows M=0..15, K = {0..7} in v0..3, {16..23} in v4..7
// lane>=16: same rows,   K = {8..15} in v0..3, {24..31} in v4..7
__device__ __forceinline__ v16bf load_a_frag(const ushort_t* A, int lda,
                                             int m0, int k0, int lane) {
  int m    = m0 + (lane & 15);
  int koff = (lane < 16) ? 0 : 8;
  const __bf16* p =
      reinterpret_cast<const __bf16*>(A + (size_t)m * lda + k0 + koff);
  v8bf lo = *reinterpret_cast<const v8bf*>(p);        // K = koff+0..7
  v8bf hi = *reinterpret_cast<const v8bf*>(p + 16);   // K = koff+16..23
  v16bf a;
#pragma unroll
  for (int i = 0; i < 8; ++i) { a[i] = lo[i]; a[i + 8] = hi[i]; }
  return a;
}

// B-matrix 32x16 bf16 from COLUMN-major storage: element (k, n) at Bc[n*ldb + k].
// lane<16: col N=lane,    K = {0..7} in v0..3, {16..23} in v4..7
// lane>=16: col N=lane-16, K = {8..15} in v0..3, {24..31} in v4..7
__device__ __forceinline__ v16bf load_b_frag_cm(const ushort_t* Bc, int ldb,
                                                int n0, int k0, int lane) {
  int n    = n0 + (lane & 15);
  int koff = (lane < 16) ? 0 : 8;
  const __bf16* p =
      reinterpret_cast<const __bf16*>(Bc + (size_t)n * ldb + k0 + koff);
  v8bf lo = *reinterpret_cast<const v8bf*>(p);
  v8bf hi = *reinterpret_cast<const v8bf*>(p + 16);
  v16bf b;
#pragma unroll
  for (int i = 0; i < 8; ++i) { b[i] = lo[i]; b[i + 8] = hi[i]; }
  return b;
}

__device__ __forceinline__ v8f wmma_bf16(v16bf a, v16bf b, v8f c) {
  return __builtin_amdgcn_wmma_f32_16x16x32_bf16(
      /*neg_a=*/false, a, /*neg_b=*/false, b,
      /*c_mod=*/(short)0, c, /*reuse_a=*/false, /*reuse_b=*/false);
}

// 2x4 register-tiled GEMM core: wave computes a 32(M) x 64(N) tile.
// 12 b128 loads feed 8 WMMAs per K-step (vs 4 loads per WMMA untiled).
__device__ __forceinline__ void gemm_2x4(const ushort_t* __restrict__ A, int lda,
                                         const ushort_t* __restrict__ Bc, int ldb,
                                         int K, int m0, int n0, int lane,
                                         v8f acc[2][4]) {
  for (int k0 = 0; k0 < K; k0 += 32) {
    v16bf a0 = load_a_frag(A, lda, m0,      k0, lane);
    v16bf a1 = load_a_frag(A, lda, m0 + 16, k0, lane);
    v16bf b0 = load_b_frag_cm(Bc, ldb, n0,      k0, lane);
    v16bf b1 = load_b_frag_cm(Bc, ldb, n0 + 16, k0, lane);
    v16bf b2 = load_b_frag_cm(Bc, ldb, n0 + 32, k0, lane);
    v16bf b3 = load_b_frag_cm(Bc, ldb, n0 + 48, k0, lane);
    acc[0][0] = wmma_bf16(a0, b0, acc[0][0]);
    acc[0][1] = wmma_bf16(a0, b1, acc[0][1]);
    acc[0][2] = wmma_bf16(a0, b2, acc[0][2]);
    acc[0][3] = wmma_bf16(a0, b3, acc[0][3]);
    acc[1][0] = wmma_bf16(a1, b0, acc[1][0]);
    acc[1][1] = wmma_bf16(a1, b1, acc[1][1]);
    acc[1][2] = wmma_bf16(a1, b2, acc[1][2]);
    acc[1][3] = wmma_bf16(a1, b3, acc[1][3]);
  }
}

// ---- Kernel 0: f32 -> bf16 weight conversion ----
__global__ void cvt_f32_bf16(const float* __restrict__ in,
                             ushort_t* __restrict__ out, int n) {
  int i = blockIdx.x * 256 + threadIdx.x;
  if (i < n) out[i] = f32_to_bf16(in[i]);
}

// ---- Kernel 1: GroupNorm, write transposed bf16 hT[b][n][c] ----
__global__ void groupnorm_bf16T(const float* __restrict__ x,
                                const float* __restrict__ gsc,
                                const float* __restrict__ gbi,
                                ushort_t* __restrict__ hT) {
  __shared__ float s1[256], s2[256];
  int blk = blockIdx.x, tid = threadIdx.x;
  int b = blk / NGRP, g = blk % NGRP;
  const float* xp = x + ((size_t)b * CCH + (size_t)g * CPG) * NSP;  // contiguous 64*1024
  const int TOT = CPG * NSP;
  float s = 0.f, ss = 0.f;
  for (int i = tid; i < TOT; i += 256) { float v = xp[i]; s += v; ss += v * v; }
  s1[tid] = s; s2[tid] = ss; __syncthreads();
  for (int st = 128; st > 0; st >>= 1) {
    if (tid < st) { s1[tid] += s1[tid + st]; s2[tid] += s2[tid + st]; }
    __syncthreads();
  }
  float mean = s1[0] * (1.0f / TOT);
  float var  = s2[0] * (1.0f / TOT) - mean * mean;
  float rstd = rsqrtf(var + 1e-5f);
  for (int i = tid; i < TOT; i += 256) {
    int cl = i >> 10, n = i & 1023;
    int c = g * CPG + cl;
    float v = (xp[i] - mean) * rstd * gsc[c] + gbi[c];
    hT[((size_t)b * NSP + n) * CCH + c] = f32_to_bf16(v);
  }
}

// ---- Kernel 2: QKV GEMM  qkv[o,n] = W[o,:]·hT[n,:] + b[o], routed to qT/kT/v ----
__global__ void qkv_gemm(const ushort_t* __restrict__ wqkv,
                         const float* __restrict__ qkv_b,
                         const ushort_t* __restrict__ hT,
                         ushort_t* __restrict__ qT, ushort_t* __restrict__ kT,
                         ushort_t* __restrict__ vb) {
  int lane = threadIdx.x;
  int b  = blockIdx.z;
  int n0 = blockIdx.x * 64;
  int m0 = (blockIdx.y * 4 + threadIdx.y) * 32;
  const ushort_t* Bm = hT + (size_t)b * NSP * CCH;
  v8f acc[2][4] = {};
  gemm_2x4(wqkv, CCH, Bm, CCH, CCH, m0, n0, lane, acc);
  int lhi = (lane < 16) ? 0 : 8;
#pragma unroll
  for (int mt = 0; mt < 2; ++mt)
#pragma unroll
    for (int nt = 0; nt < 4; ++nt) {
      int ncol  = n0 + nt * 16 + (lane & 15);
      int mbase = m0 + mt * 16 + lhi;
#pragma unroll
      for (int r = 0; r < 8; ++r) {
        int o = mbase + r;
        ushort_t bits = f32_to_bf16(acc[mt][nt][r] + qkv_b[o]);
        if (o < CCH)
          qT[((size_t)b * NSP + ncol) * CCH + o] = bits;
        else if (o < 2 * CCH)
          kT[((size_t)b * NSP + ncol) * CCH + (o - CCH)] = bits;
        else
          vb[((size_t)b * CCH + (o - 2 * CCH)) * NSP + ncol] = bits;
      }
    }
}

// ---- Kernel 3: attention scores  attn[i,j] = scale * qT[i,:]·kT[j,:] ----
__global__ void attn_gemm(const ushort_t* __restrict__ qT,
                          const ushort_t* __restrict__ kT,
                          float* __restrict__ attn) {
  const float scale = 0.044194173824159216f;  // 512^-0.5
  int lane = threadIdx.x;
  int b  = blockIdx.z;
  int j0 = blockIdx.x * 64;
  int i0 = (blockIdx.y * 4 + threadIdx.y) * 32;
  const ushort_t* A  = qT + (size_t)b * NSP * CCH;
  const ushort_t* Bm = kT + (size_t)b * NSP * CCH;
  v8f acc[2][4] = {};
  gemm_2x4(A, CCH, Bm, CCH, CCH, i0, j0, lane, acc);
  int lhi = (lane < 16) ? 0 : 8;
#pragma unroll
  for (int mt = 0; mt < 2; ++mt)
#pragma unroll
    for (int nt = 0; nt < 4; ++nt) {
      int j = j0 + nt * 16 + (lane & 15);
      int ibase = i0 + mt * 16 + lhi;
#pragma unroll
      for (int r = 0; r < 8; ++r)
        attn[((size_t)b * NSP + ibase + r) * NSP + j] = acc[mt][nt][r] * scale;
    }
}

// ---- Kernel 4: row softmax; write bf16 probs in-place in each row's slot ----
// bf16 row r lives at ushort index r*2*NSP (first half of the f32 row's bytes).
__global__ void softmax_rows(float* __restrict__ attn) {
  __shared__ float sred[256];
  int row = blockIdx.x, tid = threadIdx.x;
  float* rp = attn + (size_t)row * NSP;
  float v[4];
  float lmax = -3.4e38f;
#pragma unroll
  for (int t = 0; t < 4; ++t) { v[t] = rp[tid + t * 256]; lmax = fmaxf(lmax, v[t]); }
  sred[tid] = lmax; __syncthreads();
  for (int s = 128; s > 0; s >>= 1) {
    if (tid < s) sred[tid] = fmaxf(sred[tid], sred[tid + s]);
    __syncthreads();
  }
  float rmax = sred[0]; __syncthreads();
  float lsum = 0.f;
#pragma unroll
  for (int t = 0; t < 4; ++t) { v[t] = __expf(v[t] - rmax); lsum += v[t]; }
  sred[tid] = lsum; __syncthreads();
  for (int s = 128; s > 0; s >>= 1) {
    if (tid < s) sred[tid] += sred[tid + s];
    __syncthreads();
  }
  float inv = 1.0f / sred[0];
  ushort_t* op = reinterpret_cast<ushort_t*>(rp);  // same row only; reads are done
#pragma unroll
  for (int t = 0; t < 4; ++t) op[tid + t * 256] = f32_to_bf16(v[t] * inv);
}

// ---- Kernel 5: ao[i,c] = sum_j probs[i,j] * v[c,j] ----
__global__ void av_gemm(const ushort_t* __restrict__ attnP,  // lda = 2*NSP
                        const ushort_t* __restrict__ vb,
                        ushort_t* __restrict__ ao) {
  int lane = threadIdx.x;
  int b  = blockIdx.z;
  int c0 = blockIdx.x * 64;
  int i0 = (blockIdx.y * 4 + threadIdx.y) * 32;
  const ushort_t* A  = attnP + (size_t)b * NSP * 2 * NSP;
  const ushort_t* Bm = vb + (size_t)b * CCH * NSP;  // (k=j, col=c) at Bm[c*NSP + j]
  v8f acc[2][4] = {};
  gemm_2x4(A, 2 * NSP, Bm, NSP, NSP, i0, c0, lane, acc);
  int lhi = (lane < 16) ? 0 : 8;
#pragma unroll
  for (int mt = 0; mt < 2; ++mt)
#pragma unroll
    for (int nt = 0; nt < 4; ++nt) {
      int c = c0 + nt * 16 + (lane & 15);
      int ibase = i0 + mt * 16 + lhi;
#pragma unroll
      for (int r = 0; r < 8; ++r)
        ao[((size_t)b * NSP + ibase + r) * CCH + c] = f32_to_bf16(acc[mt][nt][r]);
    }
}

// ---- Kernel 6: proj GEMM + bias + residual -> f32 out[b][o][n] ----
__global__ void proj_gemm(const ushort_t* __restrict__ pw,
                          const float* __restrict__ proj_b,
                          const ushort_t* __restrict__ ao,  // [n][c], ld = CCH
                          const float* __restrict__ x,
                          float* __restrict__ out) {
  int lane = threadIdx.x;
  int b  = blockIdx.z;
  int n0 = blockIdx.x * 64;
  int o0 = (blockIdx.y * 4 + threadIdx.y) * 32;
  const ushort_t* Bm = ao + (size_t)b * NSP * CCH;  // (k=c, col=n) at Bm[n*CCH + c]
  v8f acc[2][4] = {};
  gemm_2x4(pw, CCH, Bm, CCH, CCH, o0, n0, lane, acc);
  int lhi = (lane < 16) ? 0 : 8;
#pragma unroll
  for (int mt = 0; mt < 2; ++mt)
#pragma unroll
    for (int nt = 0; nt < 4; ++nt) {
      int n = n0 + nt * 16 + (lane & 15);
      int obase = o0 + mt * 16 + lhi;
#pragma unroll
      for (int r = 0; r < 8; ++r) {
        int o = obase + r;
        size_t idx = ((size_t)b * CCH + o) * NSP + n;
        out[idx] = acc[mt][nt][r] + proj_b[o] + x[idx];
      }
    }
}

extern "C" void kernel_launch(void* const* d_in, const int* in_sizes, int n_in,
                              void* d_out, int out_size, void* d_ws, size_t ws_size,
                              hipStream_t stream) {
  (void)in_sizes; (void)n_in; (void)out_size; (void)ws_size;
  const float* x        = (const float*)d_in[0];
  const float* gn_scale = (const float*)d_in[1];
  const float* gn_bias  = (const float*)d_in[2];
  const float* qkv_w    = (const float*)d_in[3];
  const float* qkv_b    = (const float*)d_in[4];
  const float* proj_w   = (const float*)d_in[5];
  const float* proj_b   = (const float*)d_in[6];
  float* out = (float*)d_out;

  // Workspace layout (all offsets 256B aligned), total ~130 MB:
  char* ws = (char*)d_ws;
  ushort_t* wqkv = (ushort_t*)(ws + 0);            //  1.5 MB  bf16 qkv weights
  ushort_t* pw   = (ushort_t*)(ws + 1572864);      //  0.5 MB  bf16 proj weights
  ushort_t* hT   = (ushort_t*)(ws + 2097152);      // 16 MB    hT[b][n][c] (reused as ao)
  ushort_t* qT   = (ushort_t*)(ws + 18874368);     // 16 MB    qT[b][n][c]
  ushort_t* kT   = (ushort_t*)(ws + 35651584);     // 16 MB    kT[b][n][c]
  ushort_t* vb   = (ushort_t*)(ws + 52428800);     // 16 MB    v[b][c][n]
  float*    attn = (float*)   (ws + 69206016);     // 64 MB    scores (probs in-place)
  ushort_t* ao   = hT;                             // hT is dead after qkv_gemm

  dim3 blk128(32, 4, 1);

  // 0) weight conversion
  cvt_f32_bf16<<<(O3C * CCH + 255) / 256, 256, 0, stream>>>(qkv_w, wqkv, O3C * CCH);
  cvt_f32_bf16<<<(CCH * CCH + 255) / 256, 256, 0, stream>>>(proj_w, pw, CCH * CCH);
  // 1) GroupNorm -> hT (bf16, transposed)
  groupnorm_bf16T<<<BATCH * NGRP, 256, 0, stream>>>(x, gn_scale, gn_bias, hT);
  // 2) QKV projection: waves tile M=1536 in 32-row strips, N=1024 in 64-col strips
  qkv_gemm<<<dim3(NSP / 64, O3C / 128, BATCH), blk128, 0, stream>>>(
      wqkv, qkv_b, hT, qT, kT, vb);
  // 3) scores
  attn_gemm<<<dim3(NSP / 64, NSP / 128, BATCH), blk128, 0, stream>>>(qT, kT, attn);
  // 4) softmax (f32 -> bf16 probs in place)
  softmax_rows<<<BATCH * NSP, 256, 0, stream>>>(attn);
  // 5) probs @ V
  av_gemm<<<dim3(CCH / 64, NSP / 128, BATCH), blk128, 0, stream>>>(
      (const ushort_t*)attn, vb, ao);
  // 6) output projection + bias + residual
  proj_gemm<<<dim3(NSP / 64, CCH / 128, BATCH), blk128, 0, stream>>>(
      pw, proj_b, ao, x, out);
}
